// LambdaConv1d_54554674593926
// MI455X (gfx1250) — matrix operand order repaired and checked by reading, not compiled, not run on hardware
//
#include <hip/hip_runtime.h>
#include <hip/hip_bf16.h>
#include <cstdint>

// Problem constants (fixed by setup_inputs)
#define BB 16
#define CC 512
#define NNE 4096
#define HH 4
#define KK 16
#define UU 4
#define MME 7
#define VVD 128
#define OO 640           // 64 q rows + 64 k rows + 512 v rows
#define EPSV 1e-5f
#define LDA 40           // padded LDS row stride (bf16 elems) to spread banks
#define TN 256           // n-tile of the fused output kernel

typedef __attribute__((ext_vector_type(16))) __bf16 bf16x16;
typedef __attribute__((ext_vector_type(8)))  float  f32x8;

struct alignas(16) US8 { unsigned short u[8]; };
struct alignas(8)  US4 { unsigned short u[4]; };

union AFrag { bf16x16 v; unsigned short s[16]; US8 h[2]; };

__device__ __forceinline__ unsigned short f2bf(float f) {
  union { float f; unsigned u; } x; x.f = f;
  unsigned r = x.u + 0x7FFFu + ((x.u >> 16) & 1u);   // round-to-nearest-even
  return (unsigned short)(r >> 16);
}

// ---------------------------------------------------------------- kernel 1
// x (f32, B*C*N) -> bf16
__global__ __launch_bounds__(256) void k_convert_x(const float* __restrict__ x,
                                                   unsigned short* __restrict__ xh,
                                                   int total) {
  int i = (blockIdx.x * 256 + threadIdx.x) * 4;
  if (i + 3 >= total) return;
  float4 f = *(const float4*)(x + i);
  US4 o; o.u[0] = f2bf(f.x); o.u[1] = f2bf(f.y); o.u[2] = f2bf(f.z); o.u[3] = f2bf(f.w);
  *(US4*)(xh + i) = o;
}

// ---------------------------------------------------------------- kernel 2
// Stack [Wq*sq ; Wk ; Wv*sv] -> bf16 Wall (640x512); biasAll = [bq ; 0 ; bv]
__global__ __launch_bounds__(256) void k_prep_w(
    const float* __restrict__ Wq, const float* __restrict__ gq, const float* __restrict__ bq,
    const float* __restrict__ Wk, const float* __restrict__ Wv,
    const float* __restrict__ gv, const float* __restrict__ bv,
    unsigned short* __restrict__ Wall, float* __restrict__ biasAll) {
  int idx = blockIdx.x * 256 + threadIdx.x;
  const float inv = rsqrtf(1.0f + EPSV);   // BN: running_var = 1
  if (idx < OO * CC) {
    int o = idx / CC, c = idx - o * CC;
    float w;
    if (o < 64)        w = Wq[o * CC + c] * (gq[o] * inv);
    else if (o < 128)  w = Wk[(o - 64) * CC + c];
    else               w = Wv[(o - 128) * CC + c] * (gv[o - 128] * inv);
    Wall[idx] = f2bf(w);
  }
  if (idx < OO) {
    float bbv;
    if (idx < 64)       bbv = bq[idx];
    else if (idx < 128) bbv = 0.0f;
    else                bbv = bv[idx - 128];
    biasAll[idx] = bbv;
  }
}

// ---------------------------------------------------------------- kernel 3
// P[b] = Wall(640x512) * x_bf16[b](512x4096) + bias   (bf16 WMMA, f32 acc)
// Block: 64(M) x 128(N) tile, 8 waves, each wave 4 MxN 16x16 WMMA tiles.
__global__ __launch_bounds__(256) void k_gemm_proj(
    const unsigned short* __restrict__ xh, const unsigned short* __restrict__ Wall,
    const float* __restrict__ biasAll, float* __restrict__ P) {
  __shared__ unsigned short As[64 * LDA];    // A tile [m][k]
  __shared__ unsigned short Bs[128 * LDA];   // B tile transposed: [n][k]

  int blk = blockIdx.x;
  int b   = blk / 320;            // 10 M-tiles * 32 N-tiles per batch
  int rem = blk - b * 320;
  int ot  = rem >> 5, nt = rem & 31;
  int M0 = ot * 64, N0 = nt * 128;

  int tid  = threadIdx.x;
  int w    = tid >> 5;            // wave id (wave32)
  int lane = tid & 31;
  int half = lane >> 4, l16 = lane & 15;

  f32x8 acc[4];
#pragma unroll
  for (int i = 0; i < 4; ++i)
#pragma unroll
    for (int r = 0; r < 8; ++r) acc[i][r] = 0.0f;

  const size_t xbase = (size_t)b * CC * NNE;

  // staging thread mapping (constant across K loop)
  int am = tid >> 2, akq = (tid & 3) * 8;          // A: 64 rows x 4 chunks of 8
  int bk = tid >> 3, bn8 = (tid & 7) * 16;         // B: 32 k-rows x 8 chunks of 16 n

  for (int kc = 0; kc < CC; kc += 32) {
    __syncthreads();
    // ---- stage A (64x32 of Wall)
    *(US8*)(As + am * LDA + akq) =
        *(const US8*)(Wall + (size_t)(M0 + am) * CC + kc + akq);
    // ---- stage B (32x128 of x), transposed into Bs[n][k]
    {
      const unsigned short* src = xh + xbase + (size_t)(kc + bk) * NNE + N0 + bn8;
      if (kc + 32 < CC)
        __builtin_prefetch(src + 32 * NNE, 0, 0);  // global_prefetch_b8 next chunk
      US8 d0 = *(const US8*)(src);
      US8 d1 = *(const US8*)(src + 8);
#pragma unroll
      for (int i = 0; i < 8; ++i) Bs[(bn8 + i) * LDA + bk] = d0.u[i];
#pragma unroll
      for (int i = 0; i < 8; ++i) Bs[(bn8 + 8 + i) * LDA + bk] = d1.u[i];
    }
    __syncthreads();

    // ---- B fragment (32x16, K x N): lane holds N=w*16+l16, K kB..kB+7 and kB+16..kB+23
    AFrag bf;
    {
      int n = w * 16 + l16, kB = half * 8;
      bf.h[0] = *(const US8*)(Bs + n * LDA + kB);
      bf.h[1] = *(const US8*)(Bs + n * LDA + kB + 16);
    }
    // ---- 4 M-subtiles, A fragment per ISA 16-bit A 16x32 layout
#pragma unroll
    for (int mt = 0; mt < 4; ++mt) {
      AFrag af;
      int m = mt * 16 + l16, kA = half * 8;
      af.h[0] = *(const US8*)(As + m * LDA + kA);        // K kA..kA+7
      af.h[1] = *(const US8*)(As + m * LDA + kA + 16);   // K kA+16..kA+23
      acc[mt] = __builtin_amdgcn_wmma_f32_16x16x32_bf16(
          false, af.v, false, bf.v, (short)0, acc[mt], false, false);
    }
  }

  // epilogue: C/D layout -> VGPR r: M = r + 8*half, N = l16
  int col = N0 + w * 16 + l16;
#pragma unroll
  for (int mt = 0; mt < 4; ++mt)
#pragma unroll
    for (int r = 0; r < 8; ++r) {
      int row = M0 + mt * 16 + r + half * 8;
      P[((size_t)b * OO + row) * NNE + col] = acc[mt][r] + biasAll[row];
    }
}

// ---------------------------------------------------------------- kernel 4
// In-place softmax over N for key rows P[b, 64..127, :]
__global__ __launch_bounds__(256) void k_softmax(float* __restrict__ P) {
  int b = blockIdx.x >> 6;
  int r = blockIdx.x & 63;
  float* row = P + ((size_t)b * OO + 64 + r) * NNE;
  int tid = threadIdx.x;
  __shared__ float red[256];

  float v[16];
  float mx = -3.4e38f;
#pragma unroll
  for (int i = 0; i < 16; ++i) { v[i] = row[tid + i * 256]; mx = fmaxf(mx, v[i]); }
  red[tid] = mx; __syncthreads();
  for (int s = 128; s > 0; s >>= 1) {
    if (tid < s) red[tid] = fmaxf(red[tid], red[tid + s]);
    __syncthreads();
  }
  mx = red[0]; __syncthreads();

  float sm = 0.0f;
#pragma unroll
  for (int i = 0; i < 16; ++i) { v[i] = __expf(v[i] - mx); sm += v[i]; }
  red[tid] = sm; __syncthreads();
  for (int s = 128; s > 0; s >>= 1) {
    if (tid < s) red[tid] += red[tid + s];
    __syncthreads();
  }
  float inv = 1.0f / red[0];
#pragma unroll
  for (int i = 0; i < 16; ++i) row[tid + i * 256] = v[i] * inv;
}

// ---------------------------------------------------------------- kernel 5
// lambda_c[b,kk,vv] = sum_{uu,n} keys[b,kk,uu,n] * values[b,vv,uu,n]
// One block per batch: M=16, N=128, K=16384 bf16 WMMA GEMM (f32 staged->bf16).
__global__ __launch_bounds__(256) void k_lambda_c(const float* __restrict__ P,
                                                  float* __restrict__ lamC) {
  __shared__ unsigned short As[16 * LDA];
  __shared__ unsigned short Bs[128 * LDA];
  int b = blockIdx.x;
  int tid = threadIdx.x, w = tid >> 5, lane = tid & 31;
  int half = lane >> 4, l16 = lane & 15;

  f32x8 acc;
#pragma unroll
  for (int r = 0; r < 8; ++r) acc[r] = 0.0f;

  const size_t pb = (size_t)b * OO;
  int akk = tid >> 4, ac0 = (tid & 15) * 2;   // A staging: 2 elems/thread
  int bvv = tid >> 1, bc0 = (tid & 1) * 16;   // B staging: 16 elems/thread

  for (int it = 0; it < 512; ++it) {
    int kb = it * 32;
    int uu = kb >> 12;        // flat contraction index = uu*4096 + n
    int nn = kb & 4095;
    __syncthreads();
    {  // keys tile 16x32
      const float* src = P + (pb + 64 + akk * 4 + uu) * NNE + nn + ac0;
      As[akk * LDA + ac0]     = f2bf(src[0]);
      As[akk * LDA + ac0 + 1] = f2bf(src[1]);
    }
    {  // values tile, transposed into Bs[vv][k]
      const float* src = P + (pb + 128 + bvv * 4 + uu) * NNE + nn + bc0;
#pragma unroll
      for (int i = 0; i < 16; ++i) Bs[bvv * LDA + bc0 + i] = f2bf(src[i]);
    }
    __syncthreads();

    AFrag af, bf;
    {
      int kA = half * 8;
      af.h[0] = *(const US8*)(As + l16 * LDA + kA);
      af.h[1] = *(const US8*)(As + l16 * LDA + kA + 16);
    }
    {
      int n = w * 16 + l16, kB = half * 8;
      bf.h[0] = *(const US8*)(Bs + n * LDA + kB);
      bf.h[1] = *(const US8*)(Bs + n * LDA + kB + 16);
    }
    acc = __builtin_amdgcn_wmma_f32_16x16x32_bf16(
        false, af.v, false, bf.v, (short)0, acc, false, false);
  }
#pragma unroll
  for (int r = 0; r < 8; ++r) {
    int kk = r + half * 8, vv = w * 16 + l16;
    lamC[((size_t)b * KK + kk) * VVD + vv] = acc[r];
  }
}

// ---------------------------------------------------------------- kernel 6
// Fused: lambda_p (7-tap conv) + lambda_c, contracted with queries -> out.
// out[b, h*128+vv, n] = sum_kk q[b, h*16+kk, n] *
//                       (lamC[b,kk,vv] + sum_{uu,m} emb[kk,uu,m]*V2d[b,uu,vv,n+m-3])
__global__ __launch_bounds__(256) void k_out(const float* __restrict__ P,
                                             const float* __restrict__ lamC,
                                             const float* __restrict__ emb,
                                             float* __restrict__ out) {
  __shared__ float Vs[UU][TN + 8];
  __shared__ float Es[KK * UU * MME];      // 448 conv weights

  int b = blockIdx.x >> 4;                 // NNE / TN = 16 tiles
  int ntile = blockIdx.x & 15;
  int nbase = ntile * TN;
  int tid = threadIdx.x;
  int n = nbase + tid;
  const size_t pb = (size_t)b * OO;

  for (int j = tid; j < KK * UU * MME; j += 256) Es[j] = emb[j];

  float q[64];
#pragma unroll
  for (int r = 0; r < 64; ++r) q[r] = P[(pb + r) * NNE + n];

  for (int vv = 0; vv < VVD; ++vv) {
    __syncthreads();
    // stage 4 value rows with 3-wide halo; zero-pad outside [0,N)
    for (int j = tid; j < UU * (TN + 6); j += 256) {
      int uu = j / (TN + 6);
      int jj = j - uu * (TN + 6);
      int np = nbase + jj - 3;
      float val = 0.0f;
      if (np >= 0 && np < NNE)
        val = P[(pb + 128 + uu * VVD + vv) * NNE + np];   // values_2d[b,uu,vv,np]
      Vs[uu][jj] = val;
    }
    __syncthreads();

    float lam[16];
#pragma unroll
    for (int kk = 0; kk < 16; ++kk) lam[kk] = lamC[((size_t)b * KK + kk) * VVD + vv];
#pragma unroll
    for (int uu = 0; uu < UU; ++uu) {
      float win[7];
#pragma unroll
      for (int m = 0; m < 7; ++m) win[m] = Vs[uu][tid + m];
#pragma unroll
      for (int kk = 0; kk < 16; ++kk) {
        float a = 0.0f;
#pragma unroll
        for (int m = 0; m < 7; ++m) a = fmaf(Es[(kk * UU + uu) * MME + m], win[m], a);
        lam[kk] += a;
      }
    }
#pragma unroll
    for (int h = 0; h < HH; ++h) {
      float y = 0.0f;
#pragma unroll
      for (int kk = 0; kk < 16; ++kk) y = fmaf(q[h * 16 + kk], lam[kk], y);
      out[((size_t)b * (HH * VVD) + h * VVD + vv) * NNE + n] = y;
    }
  }
}

// ---------------------------------------------------------------- launch
extern "C" void kernel_launch(void* const* d_in, const int* in_sizes, int n_in,
                              void* d_out, int out_size, void* d_ws, size_t ws_size,
                              hipStream_t stream) {
  const float* x   = (const float*)d_in[0];
  const float* Wq  = (const float*)d_in[1];
  const float* gq  = (const float*)d_in[2];
  const float* bq  = (const float*)d_in[3];
  const float* Wk  = (const float*)d_in[4];
  const float* Wv  = (const float*)d_in[5];
  const float* gv  = (const float*)d_in[6];
  const float* bv  = (const float*)d_in[7];
  const float* emb = (const float*)d_in[8];
  float* out = (float*)d_out;

  // workspace layout (all 256B aligned)
  char* ws = (char*)d_ws;
  const size_t XH_B   = (size_t)BB * CC * NNE * 2;        //  67,108,864
  const size_t WALL_B = (size_t)OO * CC * 2;              //     655,360
  const size_t BIAS_B = (size_t)OO * 4;                   //       2,560
  const size_t P_B    = (size_t)BB * OO * NNE * 4;        // 167,772,160
  unsigned short* xh   = (unsigned short*)ws;
  unsigned short* Wall = (unsigned short*)(ws + XH_B);
  float* biasAll       = (float*)(ws + XH_B + WALL_B);
  float* P             = (float*)(ws + XH_B + WALL_B + BIAS_B);
  float* lamC          = (float*)(ws + XH_B + WALL_B + BIAS_B + P_B);
  (void)ws_size; (void)in_sizes; (void)n_in; (void)out_size;

  int total = BB * CC * NNE;
  k_convert_x<<<(total / 4 + 255) / 256, 256, 0, stream>>>(x, xh, total);
  k_prep_w<<<(OO * CC + 255) / 256, 256, 0, stream>>>(Wq, gq, bq, Wk, Wv, gv, bv,
                                                      Wall, biasAll);
  k_gemm_proj<<<BB * 10 * 32, 256, 0, stream>>>(xh, Wall, biasAll, P);
  k_softmax<<<BB * 64, 256, 0, stream>>>(P);
  k_lambda_c<<<BB, 256, 0, stream>>>(P, lamC);
  k_out<<<BB * (NNE / TN), 256, 0, stream>>>(P, lamC, emb, out);
}
